// Block_49855980372537
// MI455X (gfx1250) — matrix operand order repaired and checked by reading, not compiled, not run on hardware
//
#include <hip/hip_runtime.h>
#include <math.h>

typedef __bf16 bf16;
typedef __attribute__((ext_vector_type(16))) bf16  v16bf;
typedef __attribute__((ext_vector_type(8)))  bf16  v8bf;
typedef __attribute__((ext_vector_type(4)))  bf16  v4bf;
typedef __attribute__((ext_vector_type(8)))  float v8f;

#define WMMA_BF16(a, b, c) \
  __builtin_amdgcn_wmma_f32_16x16x32_bf16(false, (a), false, (b), (short)0, (c), false, false)

// ---------------------------------------------------------------------------
// WMMA fragment loaders (ISA 7.12.2 layouts, wave32)
// A 16x32 bf16: lanes 0-15 row M=lane hold K={k0..k0+7, k0+16..k0+23},
//               lanes 16-31 hold K={k0+8..k0+15, k0+24..k0+31}
// B 32x16 bf16 is PRE-PACKED per (kt,nt,lane): 16 contiguous bf16 (32B)
// C/D f32 16x16: lane l col (l&15); VGPR r -> row r + 8*(l>>4)
// ---------------------------------------------------------------------------
__device__ __forceinline__ v16bf lda_frag(const bf16* X, int ld, int r0, int k0, int lane) {
  int row = r0 + (lane & 15);
  int h   = (lane >> 4) << 3;       // 0 or 8
  const bf16* p1 = X + row * ld + k0 + h;
  v8bf lo = *(const v8bf*)p1;       // 16B aligned -> ds_load_b128
  v8bf hi = *(const v8bf*)(p1 + 16);
  return __builtin_shufflevector(lo, hi, 0, 1, 2, 3, 4, 5, 6, 7,
                                 8, 9, 10, 11, 12, 13, 14, 15);
}

__device__ __forceinline__ v16bf ldb_frag(const bf16* Wp, int ntiles, int kt, int nt, int lane) {
  return *(const v16bf*)(Wp + ((size_t)(kt * ntiles + nt) * 32 + lane) * 16);
}

__device__ __forceinline__ v4bf cvt4(float4 v) {
  v4bf b;
  b[0] = (bf16)v.x; b[1] = (bf16)v.y; b[2] = (bf16)v.z; b[3] = (bf16)v.w;
  return b;
}

// hidden = lrelu(X @ W1 + b1) -> LDS bf16; 4 n-tiles register-blocked per K sweep
template <int DIN, int DH>
__device__ __forceinline__ void mlp_hidden(const bf16* Xl, bf16* Hl, const bf16* W1p,
                                           const float* b1, int r0, int lane) {
  const int n15 = lane & 15, hh = lane >> 4;
#pragma unroll 1
  for (int nt0 = 0; nt0 < DH / 16; nt0 += 4) {
    v8f acc[4] = {{}, {}, {}, {}};
#pragma unroll
    for (int kt = 0; kt < DIN / 32; ++kt) {
      v16bf a = lda_frag(Xl, DIN, r0, kt * 32, lane);
#pragma unroll
      for (int i = 0; i < 4; ++i) {
        v16bf b = ldb_frag(W1p, DH / 16, kt, nt0 + i, lane);
        acc[i] = WMMA_BF16(a, b, acc[i]);
      }
    }
#pragma unroll
    for (int i = 0; i < 4; ++i) {
      int n = (nt0 + i) * 16 + n15;
      float bias = b1[n];
#pragma unroll
      for (int r = 0; r < 8; ++r) {
        float v = acc[i][r] + bias;
        v = (v >= 0.f) ? v : 0.01f * v;
        Hl[(r0 + hh * 8 + r) * DH + n] = (bf16)v;
      }
    }
  }
}

// out = H @ W2 (4 n-tiles at once, accumulators returned for custom epilogue)
template <int DH, int DOUT>
__device__ __forceinline__ void mlp_out4(const bf16* Hl, const bf16* W2p,
                                         int r0, int nt0, int lane, v8f acc[4]) {
  acc[0] = v8f{}; acc[1] = v8f{}; acc[2] = v8f{}; acc[3] = v8f{};
#pragma unroll
  for (int kt = 0; kt < DH / 32; ++kt) {
    v16bf a = lda_frag(Hl, DH, r0, kt * 32, lane);
#pragma unroll
    for (int i = 0; i < 4; ++i) {
      v16bf b = ldb_frag(W2p, DOUT / 16, kt, nt0 + i, lane);
      acc[i] = WMMA_BF16(a, b, acc[i]);
    }
  }
}

// pack W (DK x DN, row-major f32) into per-fragment contiguous bf16 layout
__global__ void k_pack_b(const float* __restrict__ W, bf16* __restrict__ Wp, int DK, int DN) {
  int t = blockIdx.x * blockDim.x + threadIdx.x;
  int ntiles = DN >> 4;
  int total = (DK >> 5) * ntiles * 32;
  if (t >= total) return;
  int lane = t & 31;
  int f = t >> 5;
  int nt = f % ntiles;
  int kt = f / ntiles;
  int n  = nt * 16 + (lane & 15);
  int kb = kt * 32 + ((lane >> 4) << 3);
  bf16* d = Wp + (size_t)t * 16;
#pragma unroll
  for (int j = 0; j < 16; ++j) d[j] = (bf16)W[(size_t)(kb + j + ((j >> 3) << 3)) * DN + n];
}

// ---------------------------------------------------------------------------
// Edge MLP: 800k x (256 -> 256 -> 64), pre-BN output + BN moment accumulation
// ---------------------------------------------------------------------------
__global__ void k_edge_mlp(const float* __restrict__ xs, const float* __restrict__ xt,
                           const int* __restrict__ src, const int* __restrict__ tgt,
                           const float* __restrict__ ea, const float* __restrict__ xu,
                           const bf16* __restrict__ W1, const float* __restrict__ b1,
                           const bf16* __restrict__ W2, const float* __restrict__ b2,
                           float* __restrict__ outE, float* __restrict__ bnacc, int E) {
  extern __shared__ char smem[];
  bf16* Xl = (bf16*)smem;                // 128x256
  bf16* Hl = Xl + 128 * 256;             // 128x256
  int*  ids = (int*)(Hl + 128 * 256);    // src[128], tgt[128]
  const int base = blockIdx.x * 128;
  const int tid  = threadIdx.x;
  if (tid < 128) {
    int e = base + tid;
    ids[tid]       = (e < E) ? src[e] : 0;
    ids[128 + tid] = (e < E) ? tgt[e] : 0;
  }
  __syncthreads();
#pragma unroll 1
  for (int q = tid; q < 128 * 64; q += 256) {     // float4 gather
    int row = q >> 6, col = (q & 63) << 2;
    int e = base + row;
    float4 v = make_float4(0.f, 0.f, 0.f, 0.f);
    if (e < E) {
      if      (col < 64)  v = *(const float4*)(xs + ids[row] * 64 + col);
      else if (col < 128) v = *(const float4*)(xt + ids[128 + row] * 64 + col - 64);
      else if (col < 192) v = *(const float4*)(ea + (size_t)e * 64 + col - 128);
      else                v = *(const float4*)(xu + col - 192);
    }
    *(v4bf*)(Xl + row * 256 + col) = cvt4(v);
  }
  __syncthreads();
  const int lane = tid & 31, r0 = (tid >> 5) * 16;
  mlp_hidden<256, 256>(Xl, Hl, W1, b1, r0, lane);
  const int n15 = lane & 15, hh = lane >> 4;
  v8f acc[4];
  mlp_out4<256, 64>(Hl, W2, r0, 0, lane, acc);
#pragma unroll
  for (int i = 0; i < 4; ++i) {
    int n = i * 16 + n15;
    float bias = b2[n];
    float s = 0.f, ss = 0.f;
#pragma unroll
    for (int r = 0; r < 8; ++r) {
      int e = base + r0 + hh * 8 + r;
      if (e < E) {
        float v = acc[i][r] + bias;
        outE[e * 64 + n] = v;
        s += v; ss += v * v;
      }
    }
    atomicAdd(&bnacc[n], s);
    atomicAdd(&bnacc[64 + n], ss);
  }
}

// ---------------------------------------------------------------------------
// Source messages: 800k x (128 -> 128 -> 128) + raw-moment scatter (L2 atomics)
// ---------------------------------------------------------------------------
__global__ void k_src_msg(const float* __restrict__ xt, const float* __restrict__ eaN,
                          const int* __restrict__ src, const int* __restrict__ tgt,
                          const bf16* __restrict__ W1, const float* __restrict__ b1,
                          const bf16* __restrict__ W2, const float* __restrict__ b2,
                          float* __restrict__ s1, float* __restrict__ s2,
                          float* __restrict__ s3, float* __restrict__ s4,
                          float* __restrict__ cnt, int E) {
  extern __shared__ char smem[];
  bf16* Xl = (bf16*)smem;                // 128x128
  bf16* Hl = Xl + 128 * 128;
  int*  ids = (int*)(Hl + 128 * 128);
  const int base = blockIdx.x * 128;
  const int tid  = threadIdx.x;
  if (tid < 128) {
    int e = base + tid;
    int sI = (e < E) ? src[e] : 0;
    ids[tid]       = sI;
    ids[128 + tid] = (e < E) ? tgt[e] : 0;
    if (e < E) atomicAdd(&cnt[sI], 1.0f);
  }
  __syncthreads();
#pragma unroll 1
  for (int q = tid; q < 128 * 32; q += 256) {
    int row = q >> 5, col = (q & 31) << 2;
    int e = base + row;
    float4 v = make_float4(0.f, 0.f, 0.f, 0.f);
    if (e < E) {
      v = (col < 64) ? *(const float4*)(xt + ids[128 + row] * 64 + col)
                     : *(const float4*)(eaN + (size_t)e * 64 + col - 64);
    }
    *(v4bf*)(Xl + row * 128 + col) = cvt4(v);
  }
  __syncthreads();
  const int lane = tid & 31, r0 = (tid >> 5) * 16;
  mlp_hidden<128, 128>(Xl, Hl, W1, b1, r0, lane);
  const int n15 = lane & 15, hh = lane >> 4;
#pragma unroll 1
  for (int nt0 = 0; nt0 < 8; nt0 += 4) {
    v8f acc[4];
    mlp_out4<128, 128>(Hl, W2, r0, nt0, lane, acc);
#pragma unroll
    for (int i = 0; i < 4; ++i) {
      int n = (nt0 + i) * 16 + n15;
      float bias = b2[n];
#pragma unroll 1
      for (int r = 0; r < 8; ++r) {
        int m = r0 + hh * 8 + r;
        int e = base + m;
        if (e < E) {
          float v = acc[i][r] + bias;
          int o = ids[m] * 128 + n;
          float v2 = v * v;
          atomicAdd(&s1[o], v);
          atomicAdd(&s2[o], v2);
          atomicAdd(&s3[o], v2 * v);
          atomicAdd(&s4[o], v2 * v2);
        }
      }
    }
  }
}

// ---------------------------------------------------------------------------
// Target messages: 800k x (128 -> 128 -> 128) + segment-sum scatter
// ---------------------------------------------------------------------------
__global__ void k_tgt_msg(const float* __restrict__ xsN, const float* __restrict__ eaN,
                          const int* __restrict__ src, const int* __restrict__ tgt,
                          const bf16* __restrict__ W1, const float* __restrict__ b1,
                          const bf16* __restrict__ W2, const float* __restrict__ b2,
                          float* __restrict__ agg, int E) {
  extern __shared__ char smem[];
  bf16* Xl = (bf16*)smem;
  bf16* Hl = Xl + 128 * 128;
  int*  ids = (int*)(Hl + 128 * 128);
  const int base = blockIdx.x * 128;
  const int tid  = threadIdx.x;
  if (tid < 128) {
    int e = base + tid;
    ids[tid]       = (e < E) ? src[e] : 0;
    ids[128 + tid] = (e < E) ? tgt[e] : 0;
  }
  __syncthreads();
#pragma unroll 1
  for (int q = tid; q < 128 * 32; q += 256) {
    int row = q >> 5, col = (q & 31) << 2;
    int e = base + row;
    float4 v = make_float4(0.f, 0.f, 0.f, 0.f);
    if (e < E) {
      v = (col < 64) ? *(const float4*)(xsN + ids[row] * 64 + col)
                     : *(const float4*)(eaN + (size_t)e * 64 + col - 64);
    }
    *(v4bf*)(Xl + row * 128 + col) = cvt4(v);
  }
  __syncthreads();
  const int lane = tid & 31, r0 = (tid >> 5) * 16;
  mlp_hidden<128, 128>(Xl, Hl, W1, b1, r0, lane);
  const int n15 = lane & 15, hh = lane >> 4;
#pragma unroll 1
  for (int nt0 = 0; nt0 < 8; nt0 += 4) {
    v8f acc[4];
    mlp_out4<128, 128>(Hl, W2, r0, nt0, lane, acc);
#pragma unroll
    for (int i = 0; i < 4; ++i) {
      int n = (nt0 + i) * 16 + n15;
      float bias = b2[n];
#pragma unroll 1
      for (int r = 0; r < 8; ++r) {
        int m = r0 + hh * 8 + r;
        int e = base + m;
        if (e < E) atomicAdd(&agg[ids[128 + m] * 128 + n], acc[i][r] + bias);
      }
    }
  }
}

// raw moments -> mean / std / skew / kurt (in place)
__global__ void k_moments(float* __restrict__ s1, float* __restrict__ s2,
                          float* __restrict__ s3, float* __restrict__ s4,
                          const float* __restrict__ cnt, int n) {
  int i = blockIdx.x * blockDim.x + threadIdx.x;
  if (i >= n) return;
  float safe = fmaxf(cnt[i >> 7], 1.0f);
  float M1 = s1[i] / safe, M2 = s2[i] / safe, M3 = s3[i] / safe, M4 = s4[i] / safe;
  float var = M2 - M1 * M1;
  var = (var >= 0.f) ? var : 0.01f * var;                    // lrelu
  float sd  = sqrtf(var + 1e-6f);
  float m3  = M3 - 3.f * M1 * M2 + 2.f * M1 * M1 * M1;
  float m4  = M4 - 4.f * M1 * M3 + 6.f * M1 * M1 * M2 - 3.f * M1 * M1 * M1 * M1;
  float sd3 = sd * sd * sd;
  s1[i] = M1; s2[i] = sd; s3[i] = m3 / sd3; s4[i] = m4 / (sd3 * sd);
}

// ---------------------------------------------------------------------------
// Source update: 25k x (640 -> 640 -> 64); 64 rows / 4 waves / block
// ---------------------------------------------------------------------------
__global__ void k_src_upd(const float* __restrict__ xs, const float* __restrict__ m1,
                          const float* __restrict__ m2, const float* __restrict__ m3,
                          const float* __restrict__ m4, const float* __restrict__ xu,
                          const bf16* __restrict__ W1, const float* __restrict__ b1,
                          const bf16* __restrict__ W2, const float* __restrict__ b2,
                          float* __restrict__ outX, float* __restrict__ bnacc, int N) {
  extern __shared__ char smem[];
  bf16* Xl = (bf16*)smem;        // 64x640
  bf16* Hl = Xl + 64 * 640;      // 64x640
  const int base = blockIdx.x * 64;
  const int tid  = threadIdx.x;  // 128 threads
#pragma unroll 1
  for (int q = tid; q < 64 * 160; q += 128) {
    int row = q / 160, col = (q - row * 160) << 2;
    int i = base + row;
    float4 v = make_float4(0.f, 0.f, 0.f, 0.f);
    if (i < N) {
      if      (col < 64)  v = *(const float4*)(xs + i * 64 + col);
      else if (col < 192) v = *(const float4*)(m1 + (size_t)i * 128 + col - 64);
      else if (col < 320) v = *(const float4*)(m2 + (size_t)i * 128 + col - 192);
      else if (col < 448) v = *(const float4*)(m3 + (size_t)i * 128 + col - 320);
      else if (col < 576) v = *(const float4*)(m4 + (size_t)i * 128 + col - 448);
      else                v = *(const float4*)(xu + col - 576);
    }
    *(v4bf*)(Xl + row * 640 + col) = cvt4(v);
  }
  __syncthreads();
  const int lane = tid & 31, r0 = (tid >> 5) * 16;
  mlp_hidden<640, 640>(Xl, Hl, W1, b1, r0, lane);
  const int n15 = lane & 15, hh = lane >> 4;
  v8f acc[4];
  mlp_out4<640, 64>(Hl, W2, r0, 0, lane, acc);
#pragma unroll
  for (int i = 0; i < 4; ++i) {
    int n = i * 16 + n15;
    float bias = b2[n];
    float s = 0.f, ss = 0.f;
#pragma unroll
    for (int r = 0; r < 8; ++r) {
      int row = base + r0 + hh * 8 + r;
      if (row < N) {
        float v = acc[i][r] + bias;
        outX[row * 64 + n] = v;
        s += v; ss += v * v;
      }
    }
    atomicAdd(&bnacc[n], s);
    atomicAdd(&bnacc[64 + n], ss);
  }
}

// ---------------------------------------------------------------------------
// Target update: 25k x (256 -> 256 -> 64)
// ---------------------------------------------------------------------------
__global__ void k_tgt_upd(const float* __restrict__ xt, const float* __restrict__ agg,
                          const float* __restrict__ xu,
                          const bf16* __restrict__ W1, const float* __restrict__ b1,
                          const bf16* __restrict__ W2, const float* __restrict__ b2,
                          float* __restrict__ outX, float* __restrict__ bnacc, int N) {
  extern __shared__ char smem[];
  bf16* Xl = (bf16*)smem;        // 128x256
  bf16* Hl = Xl + 128 * 256;
  const int base = blockIdx.x * 128;
  const int tid  = threadIdx.x;  // 256 threads
#pragma unroll 1
  for (int q = tid; q < 128 * 64; q += 256) {
    int row = q >> 6, col = (q & 63) << 2;
    int i = base + row;
    float4 v = make_float4(0.f, 0.f, 0.f, 0.f);
    if (i < N) {
      if      (col < 64)  v = *(const float4*)(xt + i * 64 + col);
      else if (col < 192) v = *(const float4*)(agg + (size_t)i * 128 + col - 64);
      else                v = *(const float4*)(xu + col - 192);
    }
    *(v4bf*)(Xl + row * 256 + col) = cvt4(v);
  }
  __syncthreads();
  const int lane = tid & 31, r0 = (tid >> 5) * 16;
  mlp_hidden<256, 256>(Xl, Hl, W1, b1, r0, lane);
  const int n15 = lane & 15, hh = lane >> 4;
  v8f acc[4];
  mlp_out4<256, 64>(Hl, W2, r0, 0, lane, acc);
#pragma unroll
  for (int i = 0; i < 4; ++i) {
    int n = i * 16 + n15;
    float bias = b2[n];
    float s = 0.f, ss = 0.f;
#pragma unroll
    for (int r = 0; r < 8; ++r) {
      int row = base + r0 + hh * 8 + r;
      if (row < N) {
        float v = acc[i][r] + bias;
        outX[row * 64 + n] = v;
        s += v; ss += v * v;
      }
    }
    atomicAdd(&bnacc[n], s);
    atomicAdd(&bnacc[64 + n], ss);
  }
}

// ---------------------------------------------------------------------------
// BN finalize (scale/shift) + in-place apply (float4)
// ---------------------------------------------------------------------------
__global__ void k_bn_fin(const float* __restrict__ acc, const float* __restrict__ g,
                         const float* __restrict__ b, float* __restrict__ scsh, float n) {
  int c = threadIdx.x;
  if (c >= 64) return;
  float m  = acc[c] / n;
  float v  = acc[64 + c] / n - m * m;
  float sc = g[c] * rsqrtf(v + 1e-5f);
  scsh[c] = sc;
  scsh[64 + c] = b[c] - m * sc;
}

__global__ void k_bn_apply(float* __restrict__ x, const float* __restrict__ scsh, int total4) {
  int i = blockIdx.x * blockDim.x + threadIdx.x;
  if (i >= total4) return;
  int c4 = i & 15;
  float4 v  = ((float4*)x)[i];
  float4 sc = ((const float4*)scsh)[c4];
  float4 sh = ((const float4*)(scsh + 64))[c4];
  v.x = v.x * sc.x + sh.x; v.y = v.y * sc.y + sh.y;
  v.z = v.z * sc.z + sh.z; v.w = v.w * sc.w + sh.w;
  ((float4*)x)[i] = v;
}

// column means of x_s (cols 0-63) and x_t (cols 64-127)
__global__ void k_colmean(const float* __restrict__ xs, const float* __restrict__ xt,
                          float* __restrict__ mean, int NS, int NT) {
  __shared__ float red[256];
  int c = blockIdx.x;
  const float* x = (c < 64) ? (xs + c) : (xt + (c - 64));
  int n = (c < 64) ? NS : NT;
  float s = 0.f;
  for (int r = threadIdx.x; r < n; r += 256) s += x[(size_t)r * 64];
  red[threadIdx.x] = s;
  __syncthreads();
  for (int st = 128; st > 0; st >>= 1) {
    if ((int)threadIdx.x < st) red[threadIdx.x] += red[threadIdx.x + st];
    __syncthreads();
  }
  if (threadIdx.x == 0) mean[c] = red[0] / (float)n;
}

// global update: 1 x (192 -> 192 -> 64) + RMSNorm (trivial, plain f32)
__global__ void k_glob(const float* __restrict__ xu, const float* __restrict__ mean,
                       const float* __restrict__ W1, const float* __restrict__ b1,
                       const float* __restrict__ W2, const float* __restrict__ b2,
                       const float* __restrict__ rmsw, float* __restrict__ out) {
  __shared__ float u[192], h[192], o[64], rms;
  int t = threadIdx.x;
  if (t < 64) u[t] = xu[t];
  else if (t < 192) u[t] = mean[t - 64];
  __syncthreads();
  if (t < 192) {
    float a = b1[t];
    for (int k = 0; k < 192; ++k) a += u[k] * W1[k * 192 + t];
    h[t] = (a >= 0.f) ? a : 0.01f * a;
  }
  __syncthreads();
  if (t < 64) {
    float a = b2[t];
    for (int k = 0; k < 192; ++k) a += h[k] * W2[k * 64 + t];
    o[t] = a;
  }
  __syncthreads();
  if (t == 0) {
    float ss = 0.f;
    for (int k = 0; k < 64; ++k) ss += o[k] * o[k];
    rms = rsqrtf(ss / 64.f + 1.1920929e-07f);
  }
  __syncthreads();
  if (t < 64) out[t] = o[t] * rms * rmsw[t];
}

// ---------------------------------------------------------------------------
extern "C" void kernel_launch(void* const* d_in, const int* in_sizes, int n_in,
                              void* d_out, int out_size, void* d_ws, size_t ws_size,
                              hipStream_t stream) {
  (void)in_sizes; (void)n_in; (void)out_size; (void)ws_size;
  const int NS = 25000, NT = 25000, E = 800000;

  const float* x_s = (const float*)d_in[0];
  const float* x_t = (const float*)d_in[1];
  const int*   ei  = (const int*)d_in[2];
  const float* ea  = (const float*)d_in[3];
  const float* x_u = (const float*)d_in[4];
  // params (insertion order of nested dicts)
  const float* eW1 = (const float*)d_in[5];  const float* eb1 = (const float*)d_in[6];
  const float* eW2 = (const float*)d_in[7];  const float* eb2 = (const float*)d_in[8];
  const float* eg  = (const float*)d_in[9];  const float* ebb = (const float*)d_in[10];
  const float* smW1= (const float*)d_in[11]; const float* smb1= (const float*)d_in[12];
  const float* smW2= (const float*)d_in[13]; const float* smb2= (const float*)d_in[14];
  const float* suW1= (const float*)d_in[15]; const float* sub1= (const float*)d_in[16];
  const float* suW2= (const float*)d_in[17]; const float* sub2= (const float*)d_in[18];
  const float* sg  = (const float*)d_in[19]; const float* sbb = (const float*)d_in[20];
  const float* tmW1= (const float*)d_in[21]; const float* tmb1= (const float*)d_in[22];
  const float* tmW2= (const float*)d_in[23]; const float* tmb2= (const float*)d_in[24];
  const float* tuW1= (const float*)d_in[25]; const float* tub1= (const float*)d_in[26];
  const float* tuW2= (const float*)d_in[27]; const float* tub2= (const float*)d_in[28];
  const float* tg  = (const float*)d_in[29]; const float* tbb = (const float*)d_in[30];
  const float* gW1 = (const float*)d_in[31]; const float* gb1 = (const float*)d_in[32];
  const float* gW2 = (const float*)d_in[33]; const float* gb2 = (const float*)d_in[34];
  const float* rmsw= (const float*)d_in[35];

  const int* srcI = ei;
  const int* tgtI = ei + E;

  float* out_xs = (float*)d_out;
  float* out_xt = out_xs + (size_t)NS * 64;
  float* out_ea = out_xt + (size_t)NT * 64;
  float* out_xu = out_ea + (size_t)E * 64;

  // workspace layout (float offsets)
  float* w = (float*)d_ws;
  float* s1   = w;                         // NS*128
  float* s2   = s1 + 3200000;
  float* s3   = s2 + 3200000;
  float* s4   = s3 + 3200000;
  float* cnt  = s4 + 3200000;              // NS
  float* agg  = cnt + 25000;               // NT*128
  float* bnE  = agg + 3200000;             // 128
  float* bnS  = bnE + 128;
  float* bnT  = bnS + 128;
  const size_t ZERO_FLOATS = 12800000 + 25000 + 3200000 + 3 * 128;
  float* scE  = bnT + 128;                 // 128 each (written before read)
  float* scS  = scE + 128;
  float* scT  = scS + 128;
  float* mean = scT + 128;                 // 128
  bf16*  WB   = (bf16*)(w + 16032000);     // packed bf16 weight area (32B aligned)
  bf16* W1e = WB;            bf16* W2e = W1e + 65536;
  bf16* W1sm= W2e + 16384;   bf16* W2sm= W1sm + 16384;
  bf16* W1su= W2sm + 16384;  bf16* W2su= W1su + 409600;
  bf16* W1tm= W2su + 40960;  bf16* W2tm= W1tm + 16384;
  bf16* W1tu= W2tm + 16384;  bf16* W2tu= W1tu + 65536;

  (void)hipMemsetAsync(d_ws, 0, ZERO_FLOATS * sizeof(float), stream);

  auto pk = [&](const float* s, bf16* d, int dk, int dn) {
    int total = (dk >> 5) * (dn >> 4) * 32;
    k_pack_b<<<(total + 255) / 256, 256, 0, stream>>>(s, d, dk, dn);
  };
  pk(eW1,  W1e,  256, 256); pk(eW2,  W2e,  256, 64);
  pk(smW1, W1sm, 128, 128); pk(smW2, W2sm, 128, 128);
  pk(suW1, W1su, 640, 640); pk(suW2, W2su, 640, 64);
  pk(tmW1, W1tm, 128, 128); pk(tmW2, W2tm, 128, 128);
  pk(tuW1, W1tu, 256, 256); pk(tuW2, W2tu, 256, 64);

  // ---- edge update ----
  k_edge_mlp<<<(E + 127) / 128, 256, 128 * 256 * 2 * 2 + 1024, stream>>>(
      x_s, x_t, srcI, tgtI, ea, x_u, W1e, eb1, W2e, eb2, out_ea, bnE, E);
  k_bn_fin<<<1, 64, 0, stream>>>(bnE, eg, ebb, scE, (float)E);
  k_bn_apply<<<(E * 16 + 255) / 256, 256, 0, stream>>>(out_ea, scE, E * 16);

  // ---- source update ----
  k_src_msg<<<(E + 127) / 128, 256, 128 * 128 * 2 * 2 + 1024, stream>>>(
      x_t, out_ea, srcI, tgtI, W1sm, smb1, W2sm, smb2, s1, s2, s3, s4, cnt, E);
  k_moments<<<(NS * 128 + 255) / 256, 256, 0, stream>>>(s1, s2, s3, s4, cnt, NS * 128);
  k_src_upd<<<(NS + 63) / 64, 128, 64 * 640 * 2 * 2, stream>>>(
      x_s, s1, s2, s3, s4, x_u, W1su, sub1, W2su, sub2, out_xs, bnS, NS);
  k_bn_fin<<<1, 64, 0, stream>>>(bnS, sg, sbb, scS, (float)NS);
  k_bn_apply<<<(NS * 16 + 255) / 256, 256, 0, stream>>>(out_xs, scS, NS * 16);

  // ---- target update ----
  k_tgt_msg<<<(E + 127) / 128, 256, 128 * 128 * 2 * 2 + 1024, stream>>>(
      out_xs, out_ea, srcI, tgtI, W1tm, tmb1, W2tm, tmb2, agg, E);
  k_tgt_upd<<<(NT + 127) / 128, 256, 128 * 256 * 2 * 2, stream>>>(
      x_t, agg, x_u, W1tu, tub1, W2tu, tub2, out_xt, bnT, NT);
  k_bn_fin<<<1, 64, 0, stream>>>(bnT, tg, tbb, scT, (float)NT);
  k_bn_apply<<<(NT * 16 + 255) / 256, 256, 0, stream>>>(out_xt, scT, NT * 16);

  // ---- global update ----
  k_colmean<<<128, 256, 0, stream>>>(out_xs, out_xt, mean, NS, NT);
  k_glob<<<1, 192, 0, stream>>>(x_u, mean, gW1, gb1, gW2, gb2, rmsw, out_xu);
}
